// GaussianRasterizer_78572131713310
// MI455X (gfx1250) — matrix operand order repaired and checked by reading, not compiled, not run on hardware
//
#include <hip/hip_runtime.h>
#include <hip/hip_bf16.h>

#define HH 192
#define WW 192
#define HWTOT (HH*WW)
#define NG 1024
#define TANFOVX 0.5f
#define TANFOVY 0.5f
#define SCALE_MOD 1.0f
#define FXc ((float)WW / (2.0f * TANFOVX))
#define FYc ((float)HH / (2.0f * TANFOVY))

typedef __attribute__((ext_vector_type(16))) _Float16 v16h;
typedef __attribute__((ext_vector_type(8)))  float    v8f;
typedef __attribute__((ext_vector_type(4)))  unsigned int u32x4;
typedef __attribute__((ext_vector_type(4)))  float    f32x4;

struct A8 { u32x4 lo; u32x4 hi; };   // 32 bytes -> bit_cast to v16h

static __device__ inline unsigned int pkf16(float a, float b) {
  _Float16 ha = (_Float16)a, hb = (_Float16)b;
  unsigned short ua = __builtin_bit_cast(unsigned short, ha);
  unsigned short ub = __builtin_bit_cast(unsigned short, hb);
  return (unsigned int)ua | ((unsigned int)ub << 16);
}

// ---------------- Kernel 1: per-gaussian preprocess ----------------
__global__ void k_preprocess(const float* __restrict__ means3D,
                             const float* __restrict__ colors,
                             const float* __restrict__ opac,
                             const float* __restrict__ scales,
                             const float* __restrict__ rots,
                             const float* __restrict__ viewm,
                             const float* __restrict__ projm,
                             float* __restrict__ params,
                             float* __restrict__ depthArr,
                             int* __restrict__ radiiOut) {
  int i = blockIdx.x * blockDim.x + threadIdx.x;
  if (i >= NG) return;

  float ph[4] = { means3D[i*3+0], means3D[i*3+1], means3D[i*3+2], 1.0f };
  float pv[4], pm[4];
  #pragma unroll
  for (int j = 0; j < 4; ++j) {
    float a = 0.f, b = 0.f;
    #pragma unroll
    for (int k = 0; k < 4; ++k) { a += ph[k] * viewm[k*4+j]; b += ph[k] * projm[k*4+j]; }
    pv[j] = a; pm[j] = b;
  }
  float depth = pv[2];
  float p_w   = 1.0f / (pm[3] + 1e-7f);
  float ndcx  = pm[0] * p_w, ndcy = pm[1] * p_w;
  float pixx  = ((ndcx + 1.0f) * (float)WW - 1.0f) * 0.5f;
  float pixy  = ((ndcy + 1.0f) * (float)HH - 1.0f) * 0.5f;

  float qr = rots[i*4+0], qx = rots[i*4+1], qy = rots[i*4+2], qz = rots[i*4+3];
  float R[3][3];
  R[0][0] = 1.f - 2.f*(qy*qy + qz*qz); R[0][1] = 2.f*(qx*qy - qr*qz); R[0][2] = 2.f*(qx*qz + qr*qy);
  R[1][0] = 2.f*(qx*qy + qr*qz); R[1][1] = 1.f - 2.f*(qx*qx + qz*qz); R[1][2] = 2.f*(qy*qz - qr*qx);
  R[2][0] = 2.f*(qx*qz - qr*qy); R[2][1] = 2.f*(qy*qz + qr*qx); R[2][2] = 1.f - 2.f*(qx*qx + qy*qy);
  float s[3] = { scales[i*3+0]*SCALE_MOD, scales[i*3+1]*SCALE_MOD, scales[i*3+2]*SCALE_MOD };
  float Mm[3][3];
  #pragma unroll
  for (int a = 0; a < 3; ++a)
    #pragma unroll
    for (int b = 0; b < 3; ++b) Mm[a][b] = R[a][b] * s[b];
  float Sig[3][3];
  #pragma unroll
  for (int a = 0; a < 3; ++a)
    #pragma unroll
    for (int b = 0; b < 3; ++b) {
      float acc = 0.f;
      #pragma unroll
      for (int j = 0; j < 3; ++j) acc += Mm[a][j] * Mm[b][j];
      Sig[a][b] = acc;
    }

  float tz = depth;
  float limx = 1.3f * TANFOVX, limy = 1.3f * TANFOVY;
  float txv = fminf(fmaxf(pv[0]/tz, -limx), limx) * tz;
  float tyv = fminf(fmaxf(pv[1]/tz, -limy), limy) * tz;
  float J[2][3] = { { FXc/tz, 0.f, -FXc*txv/(tz*tz) },
                    { 0.f, FYc/tz, -FYc*tyv/(tz*tz) } };
  float T[2][3];
  #pragma unroll
  for (int a = 0; a < 2; ++a)
    #pragma unroll
    for (int k = 0; k < 3; ++k) {
      float acc = 0.f;
      #pragma unroll
      for (int j = 0; j < 3; ++j) acc += J[a][j] * viewm[k*4+j];  // Wv[j][k] = V[k][j]
      T[a][k] = acc;
    }
  float U[2][3];
  #pragma unroll
  for (int a = 0; a < 2; ++a)
    #pragma unroll
    for (int k = 0; k < 3; ++k) {
      float acc = 0.f;
      #pragma unroll
      for (int j = 0; j < 3; ++j) acc += T[a][j] * Sig[j][k];
      U[a][k] = acc;
    }
  float cov00 = 0.f, cov01 = 0.f, cov11 = 0.f;
  #pragma unroll
  for (int k = 0; k < 3; ++k) {
    cov00 += U[0][k]*T[0][k]; cov01 += U[0][k]*T[1][k]; cov11 += U[1][k]*T[1][k];
  }
  float a2 = cov00 + 0.3f, c2 = cov11 + 0.3f, b2 = cov01;
  float det = a2*c2 - b2*b2;
  float det_inv = (det != 0.f) ? (1.f/det) : 0.f;
  float conA =  c2 * det_inv;
  float conB = -b2 * det_inv;
  float conC =  a2 * det_inv;
  float mid  = 0.5f * (a2 + c2);
  float disc = sqrtf(fmaxf(0.1f, mid*mid - det));
  float lam  = fmaxf(mid + disc, mid - disc);
  float radius = ceilf(3.0f * sqrtf(fmaxf(lam, 0.f)));
  bool valid = (depth > 0.2f) && (det > 0.f) && (radius > 0.f);

  radiiOut[i] = valid ? (int)radius : 0;
  depthArr[i] = depth;
  float* p = params + i*12;
  p[0] = pixx; p[1] = pixy; p[2] = conA; p[3] = conB; p[4] = conC;
  p[5] = valid ? opac[i] : 0.f;
  p[6] = colors[i*3+0]; p[7] = colors[i*3+1]; p[8] = colors[i*3+2];
  p[9] = depth; p[10] = 0.f; p[11] = 0.f;
}

// ---------------- Kernel 2: bitonic argsort by (depth, idx) ----------------
__global__ void __launch_bounds__(1024) k_sort(const float* __restrict__ depth,
                                               int* __restrict__ order) {
  __shared__ float kk[NG];
  __shared__ int   vv[NG];
  int t = threadIdx.x;
  kk[t] = depth[t]; vv[t] = t;
  __syncthreads();
  for (int k = 2; k <= NG; k <<= 1) {
    for (int j = k >> 1; j > 0; j >>= 1) {
      int ixj = t ^ j;
      if (ixj > t) {
        bool up = ((t & k) == 0);
        float ka = kk[t], kb = kk[ixj];
        int   va = vv[t], vb = vv[ixj];
        bool agtb = (ka > kb) || (ka == kb && va > vb);
        if (agtb == up) { kk[t] = kb; kk[ixj] = ka; vv[t] = vb; vv[ixj] = va; }
      }
      __syncthreads();
    }
  }
  order[t] = vv[t];
}

// ---------------- Kernel 3: gather params into sorted order ----------------
__global__ void k_gather(const float* __restrict__ params,
                         const int* __restrict__ order,
                         float* __restrict__ sp) {
  int j = blockIdx.x * blockDim.x + threadIdx.x;
  if (j >= NG) return;
  int o = order[j];
  #pragma unroll
  for (int q = 0; q < 12; ++q) sp[j*12+q] = params[o*12+q];
}

// ---- Kernel 4: pre-swizzle colors into WMMA B-matrix (32x16 f16) layout ----
// B[k][n]: VGPR v, lane l<16 -> N=l, K=(v>>2)*16+2(v&3),+1 ; lane l>=16 -> N=l-16, K=(v>>2)*16+8+2(v&3),+1
// stored as Bbuf[chunk*256 + lane*8 + v] so the render wave loads 2x b128/lane.
__global__ void k_buildB(const float* __restrict__ sp, unsigned int* __restrict__ Bbuf) {
  int t = blockIdx.x * blockDim.x + threadIdx.x;   // 0..8191
  int c = t >> 8, rem = t & 255, l = rem >> 3, v = rem & 7;
  int ch = l & 15;
  int k0 = c*32 + (v >> 2)*16 + ((l >= 16) ? 8 : 0) + 2*(v & 3);
  float a = (ch < 3) ? sp[k0*12 + 6 + ch] : 0.f;
  float b = (ch < 3) ? sp[(k0+1)*12 + 6 + ch] : 0.f;
  Bbuf[c*256 + l*8 + v] = pkf16(a, b);
}

// ---------------- Kernel 5: fused render (serial alpha loop + WMMA reduce) --
__global__ void __launch_bounds__(192) k_render(const float* __restrict__ sp,
                                                const unsigned int* __restrict__ Bbuf,
                                                const float* __restrict__ bg,
                                                float* __restrict__ out) {
  __shared__ __align__(16) unsigned int wbuf[6 * 512];  // 6 waves x 32 rows x 16 dwords
  __shared__ float tbuf[192];

  const int y    = blockIdx.x;
  const int tid  = threadIdx.x;
  const int wave = tid >> 5;
  const int lane = tid & 31;
  const int x0   = wave * 32;
  const float gx = (float)(x0 + lane);
  const float gy = (float)y;
  unsigned int* wrow = wbuf + wave * 512;

  const f32x4* sp4 = (const f32x4*)sp;
  float T = 1.0f;
  v8f acc0 = {0.f,0.f,0.f,0.f,0.f,0.f,0.f,0.f};
  v8f acc1 = {0.f,0.f,0.f,0.f,0.f,0.f,0.f,0.f};

  for (int c = 0; c < 32; ++c) {
    if (c + 1 < 32) {                       // gfx1250 global_prefetch_b8
      __builtin_prefetch(&sp[(c+1)*32*12], 0, 1);
      __builtin_prefetch(&Bbuf[(c+1)*256], 0, 1);
    }
    unsigned int wpk[16];
    #pragma unroll
    for (int k = 0; k < 32; k += 2) {
      float w01[2];
      #pragma unroll
      for (int h = 0; h < 2; ++h) {
        int g = c*32 + k + h;
        f32x4 g0 = sp4[g*3 + 0];            // px py conA conB
        f32x4 g1 = sp4[g*3 + 1];            // conC op  cr  cg
        float dx = g0.x - gx, dy = g0.y - gy;
        float power = -0.5f * (g0.z*dx*dx + g1.x*dy*dy) - g0.w*dx*dy;
        float al = fminf(0.99f, g1.y * __expf(fminf(power, 0.f)));
        float alm = (power <= 0.f && al >= (1.0f/255.0f)) ? al : 0.f;
        w01[h] = alm * T;
        T *= (1.0f - alm);
      }
      wpk[k >> 1] = pkf16(w01[0], w01[1]);
    }
    // lane's 32 weights (f16) -> LDS row `lane` (64B)
    u32x4* dst = (u32x4*)(wrow + lane * 16);
    #pragma unroll
    for (int q = 0; q < 4; ++q) {
      u32x4 t4; t4.x = wpk[q*4+0]; t4.y = wpk[q*4+1]; t4.z = wpk[q*4+2]; t4.w = wpk[q*4+3];
      dst[q] = t4;
    }
    __builtin_amdgcn_wave_barrier();
    asm volatile("s_wait_dscnt 0" ::: "memory");

    // A-matrix 16x32 f16 layout: lanes 0-15 K=0-7,16-23; lanes 16-31 K=8-15,24-31
    int r   = lane & 15;
    int off = (lane >> 4) * 4;              // dwords
    u32x4 lo0 = *(const u32x4*)(wrow + r*16       + off);
    u32x4 hi0 = *(const u32x4*)(wrow + r*16 + 8   + off);
    u32x4 lo1 = *(const u32x4*)(wrow + (16+r)*16     + off);
    u32x4 hi1 = *(const u32x4*)(wrow + (16+r)*16 + 8 + off);
    A8 s0; s0.lo = lo0; s0.hi = hi0;
    A8 s1; s1.lo = lo1; s1.hi = hi1;
    v16h A0 = __builtin_bit_cast(v16h, s0);
    v16h A1 = __builtin_bit_cast(v16h, s1);

    u32x4 b0 = *(const u32x4*)(Bbuf + c*256 + lane*8);
    u32x4 b1 = *(const u32x4*)(Bbuf + c*256 + lane*8 + 4);
    A8 sb; sb.lo = b0; sb.hi = b1;
    v16h Bm = __builtin_bit_cast(v16h, sb);

    acc0 = __builtin_amdgcn_wmma_f32_16x16x32_f16(false, A0, false, Bm, (short)0, acc0, false, false);
    acc1 = __builtin_amdgcn_wmma_f32_16x16x32_f16(false, A1, false, Bm, (short)0, acc1, false, false);
    asm volatile("" ::: "memory");
  }

  tbuf[tid] = T;
  __builtin_amdgcn_wave_barrier();
  asm volatile("s_wait_dscnt 0" ::: "memory");

  // C/D layout: lane<16 -> N=lane (channel), VGPR j -> M=j ; lane>=16 -> M=8+j
  int ch = lane & 15;
  if (ch < 3) {
    float bgc = bg[ch];
    int mb = (lane >> 4) * 8;
    #pragma unroll
    for (int j = 0; j < 8; ++j) {
      int m = mb + j;
      float Tf0 = tbuf[wave*32 + m];
      float Tf1 = tbuf[wave*32 + 16 + m];
      out[ch*HWTOT + y*WW + x0 + m]      = acc0[j] + Tf0 * bgc;
      out[ch*HWTOT + y*WW + x0 + 16 + m] = acc1[j] + Tf1 * bgc;
    }
  }
}

extern "C" void kernel_launch(void* const* d_in, const int* in_sizes, int n_in,
                              void* d_out, int out_size, void* d_ws, size_t ws_size,
                              hipStream_t stream) {
  const float* means3D = (const float*)d_in[0];
  const float* colors  = (const float*)d_in[2];
  const float* opac    = (const float*)d_in[3];
  const float* scales  = (const float*)d_in[4];
  const float* rots    = (const float*)d_in[5];
  const float* viewm   = (const float*)d_in[6];
  const float* projm   = (const float*)d_in[7];
  const float* bg      = (const float*)d_in[8];

  float* ws = (float*)d_ws;
  float*        depthA  = ws;                                 // 1024 f
  int*          order   = (int*)(ws + 1024);                  // 1024 i
  float*        params  = ws + 2048;                          // 1024*12 f
  float*        sparams = ws + 2048 + 12*1024;                // 1024*12 f
  unsigned int* Bbuf    = (unsigned int*)(ws + 2048 + 24*1024); // 8192 u32

  float* outColor = (float*)d_out;
  int*   radii    = ((int*)d_out) + HWTOT * 3;

  k_preprocess<<<4, 256, 0, stream>>>(means3D, colors, opac, scales, rots,
                                      viewm, projm, params, depthA, radii);
  k_sort<<<1, 1024, 0, stream>>>(depthA, order);
  k_gather<<<4, 256, 0, stream>>>(params, order, sparams);
  k_buildB<<<32, 256, 0, stream>>>(sparams, Bbuf);
  k_render<<<HH, 192, 0, stream>>>(sparams, Bbuf, bg, outColor);
}
